// SVDCrossVolume_42219528520135
// MI455X (gfx1250) — matrix operand order repaired and checked by reading, not compiled, not run on hardware
//
#include <hip/hip_runtime.h>

typedef __attribute__((ext_vector_type(16))) _Float16 v16h;
typedef __attribute__((ext_vector_type(8)))  _Float16 v8h;
typedef __attribute__((ext_vector_type(8)))  float    v8f;
typedef __attribute__((ext_vector_type(4)))  unsigned int su4;
typedef __attribute__((ext_vector_type(8)))  unsigned int su8;

#define BQ 2
#define NQ 8192
#define HH 128
#define WW 128
#define HWSZ (HH*WW)
#define KNN 16
#define CRGB 64
#define CLID 64
#define C0 134          /* 3+3+64+64 input channels */
#define K0PAD 160       /* 134 padded to multiple of 32 */
#define N1 128
#define N2 64
#define N3 64
#define EPSB 1e-5f

// ---------------------------------------------------------------------------
// TDM: DMA a 1-D run of f16 elements from global memory into LDS.
// D# built per CDNA5 ISA ch.8: group0 = {count=1, lds_addr, global_addr,
// type=2}; group1 = {data_size=2B, tensor_dim0=tile_dim0=stride=E}.
// Issued from one wave only (TDM ignores EXEC -> exactly one DMA op).
// ---------------------------------------------------------------------------
__device__ inline void tdm_load_weights(const _Float16* gsrc, _Float16* lds_dst,
                                        unsigned int elems) {
  unsigned long long ga = (unsigned long long)(uintptr_t)gsrc;
  unsigned int lo = (unsigned int)(uintptr_t)lds_dst;  // addr[31:0] = LDS byte address
  su4 g0;
  g0[0] = 1u;                                           // count=1, user descriptor
  g0[1] = lo;                                           // lds_addr
  g0[2] = (unsigned int)(ga & 0xffffffffu);             // global_addr[31:0]
  g0[3] = (unsigned int)((ga >> 32) & 0x01ffffffu)      // global_addr[56:32]
        | (2u << 30);                                   // type=2 ("image")
  su8 g1;
  g1[0] = (1u << 16);                                   // data_size=1 (2 bytes)
  g1[1] = (elems & 0xffffu) << 16;                      // tensor_dim0[15:0]
  g1[2] = (elems >> 16) & 0xffffu;                      // tensor_dim0[31:16]
  g1[3] = (elems & 0xffffu) << 16;                      // tile_dim0 = E (<=65535)
  g1[4] = 0u;                                           // tile_dim1/2 unused
  g1[5] = elems;                                        // tensor_dim0_stride[31:0]
  g1[6] = 0u;
  g1[7] = 0u;
  su4 gz = {0u, 0u, 0u, 0u};
  asm volatile("tensor_load_to_lds %0, %1, %2, %3"
               :: "s"(g0), "s"(g1), "s"(gz), "s"(gz)
               : "memory");
  __builtin_amdgcn_s_wait_tensorcnt(0);
}

// ---------------------------------------------------------------------------
// Weight folding: y = relu(s*(Wx+b-mu)+beta), s = gamma*rsqrt(var+eps)
// conv2d path -> f16 weights stored in WMMA B-fragment order:
//   dst[(kt*NT + nt)*512 + lane*16 + j] = W'[k = kt*32 + (lane>>4)*16 + j]
//                                           [n = nt*16 + (lane&15)]
// so each lane's v16h fragment is one contiguous, 32B-aligned chunk.
// ---------------------------------------------------------------------------
__global__ void fold2d_kernel(const float* __restrict__ W, const float* __restrict__ bb,
                              const float* __restrict__ g, const float* __restrict__ be,
                              const float* __restrict__ mu, const float* __restrict__ var,
                              int Cin, int Kpad, int Nout,
                              _Float16* __restrict__ Wf, float* __restrict__ bf) {
  int d = blockIdx.x * blockDim.x + threadIdx.x;
  if (d >= Kpad * Nout) return;
  int j    = d & 15;
  int lane = (d >> 4) & 31;
  int fr   = d >> 9;                 // fragment index = kt*NT + nt
  int NT   = Nout >> 4;
  int nt   = fr % NT;
  int kt   = fr / NT;
  int k = kt * 32 + (lane >> 4) * 16 + j;
  int n = nt * 16 + (lane & 15);
  float s = g[n] * rsqrtf(var[n] + EPSB);
  Wf[d] = (k < Cin) ? (_Float16)(s * W[n * Cin + k]) : (_Float16)0.f;
  if (d < Nout) {
    float sd = g[d] * rsqrtf(var[d] + EPSB);
    bf[d] = sd * (bb[d] - mu[d]) + be[d];
  }
}

__global__ void fold1d_kernel(const float* __restrict__ W, const float* __restrict__ bb,
                              const float* __restrict__ g, const float* __restrict__ be,
                              const float* __restrict__ mu, const float* __restrict__ var,
                              int Cin, int Nout, int hasbn,
                              float* __restrict__ Wf, float* __restrict__ bf) {
  int i = blockIdx.x * blockDim.x + threadIdx.x;
  if (i >= Cin * Nout) return;
  int n = i / Cin, k = i % Cin;
  float s = hasbn ? g[n] * rsqrtf(var[n] + EPSB) : 1.f;
  Wf[i] = s * W[i];
  if (k == 0) bf[n] = hasbn ? (s * (bb[n] - mu[n]) + be[n]) : bb[n];
}

// ---------------------------------------------------------------------------
// Brute-force KNN: one thread per query, register compare-swap top-16.
// Ref set = 16384 pts * 12B = 192KB/batch -> L2-resident.
// ---------------------------------------------------------------------------
__global__ void knn_kernel(const float* __restrict__ xyz, const float* __restrict__ refi,
                           int* __restrict__ idx_out) {
  int t = blockIdx.x * blockDim.x + threadIdx.x;
  if (t >= BQ * NQ) return;
  int b = t / NQ;
  float qx = xyz[t * 3 + 0], qy = xyz[t * 3 + 1], qz = xyz[t * 3 + 2];
  const float* rx = refi + (size_t)b * 3 * HWSZ;
  const float* ry = rx + HWSZ;
  const float* rz = ry + HWSZ;
  float bd[KNN]; int bi[KNN];
#pragma unroll
  for (int j = 0; j < KNN; ++j) { bd[j] = 3.4e38f; bi[j] = 0; }
  for (int m = 0; m < HWSZ; ++m) {
    float dx = qx - rx[m], dy = qy - ry[m], dz = qz - rz[m];
    float d = dx * dx + dy * dy + dz * dz;
    if (d < bd[KNN - 1]) {
      float cd = d; int ci = m;
#pragma unroll
      for (int j = 0; j < KNN; ++j) {
        bool sw = cd < bd[j];
        float td = sw ? bd[j] : cd; int ti = sw ? bi[j] : ci;
        bd[j] = sw ? cd : bd[j];    bi[j] = sw ? ci : bi[j];
        cd = td; ci = ti;
      }
    }
  }
#pragma unroll
  for (int j = 0; j < KNN; ++j) idx_out[(size_t)t * KNN + j] = bi[j];
}

// ---------------------------------------------------------------------------
// Feature build -> f16 X0 [B*N*K, K0PAD]; cols 134..159 zero-padded.
// ---------------------------------------------------------------------------
__global__ void feat_kernel(const float* __restrict__ xyz, const float* __restrict__ pts,
                            const float* __restrict__ rf3, const float* __restrict__ refi,
                            const float* __restrict__ lz, const int* __restrict__ idx,
                            _Float16* __restrict__ X0) {
  int r = blockIdx.x * blockDim.x + threadIdx.x;
  if (r >= BQ * NQ * KNN) return;
  int t = r / KNN;          // b*N + n
  int b = t / NQ;
  float z = lz[t];
  _Float16* xr = X0 + (size_t)r * K0PAD;
  xr[0] = (_Float16)(xyz[t * 3 + 0] * z);
  xr[1] = (_Float16)(xyz[t * 3 + 1] * z);
  xr[2] = (_Float16)(xyz[t * 3 + 2] * z);
  int m = idx[r];
  const float* ri = refi + (size_t)b * 3 * HWSZ;
  xr[3] = (_Float16)ri[0 * HWSZ + m];
  xr[4] = (_Float16)ri[1 * HWSZ + m];
  xr[5] = (_Float16)ri[2 * HWSZ + m];
  const float* wp = pts + (size_t)t * CLID;
  for (int c = 0; c < CLID; ++c) xr[6 + c] = (_Float16)wp[c];
  const float* rp = rf3 + (size_t)b * CRGB * HWSZ;
  for (int c = 0; c < CRGB; ++c) xr[6 + CLID + c] = (_Float16)rp[(size_t)c * HWSZ + m];
  for (int c = C0; c < K0PAD; ++c) xr[c] = (_Float16)0.f;
}

// ---------------------------------------------------------------------------
// WMMA GEMM + bias + ReLU.  Block = 256 threads = 8 waves; each wave owns a
// 16-row strip of a 128-row tile.  Weights DMA'd to LDS by the TDM (fragment-
// ordered, so each lane reads a contiguous v16h).  A streamed from global with
// vector loads + prefetch.  A/B fragments use contiguous k = 16*(lane/16) + s (wave32).
// ---------------------------------------------------------------------------
template <int KPAD, int NOUT, bool OUT_F32>
__global__ __launch_bounds__(256)
void gemm_bn_relu_kernel(const _Float16* __restrict__ X,
                         const _Float16* __restrict__ Wf,
                         const float* __restrict__ bias,
                         _Float16* __restrict__ Yh,
                         float* __restrict__ Yf) {
  constexpr int NT = NOUT / 16;
  __shared__ _Float16 sW[KPAD * NOUT];
  if (threadIdx.x == 0)
    tdm_load_weights(Wf, sW, (unsigned int)(KPAD * NOUT));  // one TDM op, wave 0
  __syncthreads();

  const int lane = threadIdx.x & 31;
  const int wave = threadIdx.x >> 5;
  const int hlf  = lane >> 4;      // lane group: 0 -> K/M low half, 1 -> high
  const int l16  = lane & 15;
  const int rowA = blockIdx.x * 128 + wave * 16 + l16;   // A-matrix row (M = lane%16)
  const _Float16* __restrict__ xrow = X + (size_t)rowA * KPAD;

#pragma unroll
  for (int nt = 0; nt < NT; ++nt) {
    v8f acc = {};
#pragma unroll
    for (int k0 = 0; k0 < KPAD; k0 += 32) {
      __builtin_prefetch((const void*)(xrow + k0 + 32), 0, 0);  // global_prefetch_b8
      const int ca = k0 + hlf * 16;  // A: lanes0-15 K=[0..15], lanes16-31 K=[16..31]
      v16h a = *(const v16h*)(xrow + ca);
      const int kt = k0 >> 5;        // B fragment: contiguous 32B per lane
      v16h bfrag = *(const v16h*)(sW + (((kt * NT + nt) * 32) + lane) * 16);
      acc = __builtin_amdgcn_wmma_f32_16x16x32_f16(false, a, false, bfrag,
                                                   (short)0, acc, false, false);
    }
    const int ncol = nt * 16 + l16;
    const float bn = bias[ncol];
#pragma unroll
    for (int r = 0; r < 8; ++r) {     // D: M = (lane/16)*8 + r
      const int mrow = blockIdx.x * 128 + wave * 16 + hlf * 8 + r;
      float v = acc[r] + bn;
      v = v > 0.f ? v : 0.f;
      if constexpr (OUT_F32) Yf[(size_t)mrow * NOUT + ncol] = v;
      else                   Yh[(size_t)mrow * NOUT + ncol] = (_Float16)v;
    }
  }
}

// ---------------------------------------------------------------------------
// Attention: aw = softmax_k(max_c feats), corres = sum aw*qi_xyz, x = sum aw*f
// ---------------------------------------------------------------------------
__global__ void attn_kernel(const float* __restrict__ Y3, const float* __restrict__ refi,
                            const int* __restrict__ idx,
                            float* __restrict__ corres, float* __restrict__ xfeat) {
  int t = blockIdx.x * blockDim.x + threadIdx.x;
  if (t >= BQ * NQ) return;
  int b = t / NQ;
  const float* F = Y3 + (size_t)t * KNN * N3;
  float mk[KNN];
  float mmax = -3.4e38f;
#pragma unroll
  for (int k = 0; k < KNN; ++k) {
    float mv = -3.4e38f;
    for (int c = 0; c < N3; ++c) mv = fmaxf(mv, F[k * N3 + c]);
    mk[k] = mv;
    mmax = fmaxf(mmax, mv);
  }
  float s = 0.f;
#pragma unroll
  for (int k = 0; k < KNN; ++k) { mk[k] = __expf(mk[k] - mmax); s += mk[k]; }
  float inv = 1.f / s;
  const float* ri = refi + (size_t)b * 3 * HWSZ;
  float cx = 0.f, cy = 0.f, cz = 0.f;
#pragma unroll
  for (int k = 0; k < KNN; ++k) {
    float a = mk[k] * inv;
    int m = idx[(size_t)t * KNN + k];
    cx += a * ri[0 * HWSZ + m];
    cy += a * ri[1 * HWSZ + m];
    cz += a * ri[2 * HWSZ + m];
  }
  corres[t * 3 + 0] = cx; corres[t * 3 + 1] = cy; corres[t * 3 + 2] = cz;
  for (int c = 0; c < N3; ++c) {
    float acc = 0.f;
#pragma unroll
    for (int k = 0; k < KNN; ++k) acc += mk[k] * inv * F[k * N3 + c];
    xfeat[(size_t)t * N3 + c] = acc;
  }
}

// ---------------------------------------------------------------------------
// conv1d MLP: 64 -> 64(bn,relu) -> 128(bn,relu) -> 2  (BN pre-folded)
// ---------------------------------------------------------------------------
__global__ void mlp4_kernel(const float* __restrict__ xfeat,
                            const float* __restrict__ W1, const float* __restrict__ b1,
                            const float* __restrict__ W2, const float* __restrict__ b2,
                            const float* __restrict__ W3, const float* __restrict__ b3,
                            float* __restrict__ out_w) {
  int t = blockIdx.x * blockDim.x + threadIdx.x;
  if (t >= BQ * NQ) return;
  float h0[64], h1[64], h2[128];
  for (int c = 0; c < 64; ++c) h0[c] = xfeat[(size_t)t * 64 + c];
  for (int n = 0; n < 64; ++n) {
    float a = b1[n];
    for (int k = 0; k < 64; ++k) a += W1[n * 64 + k] * h0[k];
    h1[n] = a > 0.f ? a : 0.f;
  }
  for (int n = 0; n < 128; ++n) {
    float a = b2[n];
    for (int k = 0; k < 64; ++k) a += W2[n * 64 + k] * h1[k];
    h2[n] = a > 0.f ? a : 0.f;
  }
  for (int n = 0; n < 2; ++n) {
    float a = b3[n];
    for (int k = 0; k < 128; ++k) a += W3[n * 128 + k] * h2[k];
    out_w[(size_t)t * 2 + n] = a;
  }
}

// ---------------------------------------------------------------------------
// PnP: accumulate M = sum_n w^2 (r1 r1^T + r2 r2^T) (upper triangle, 78 vals)
// ---------------------------------------------------------------------------
__global__ void pnp_accum_kernel(const float* __restrict__ xyz, const float* __restrict__ lz,
                                 const float* __restrict__ corres, const float* __restrict__ gt,
                                 float* __restrict__ Macc) {
  int b = blockIdx.x;
  int tid = threadIdx.x;
  float acc[78];
#pragma unroll
  for (int q = 0; q < 78; ++q) acc[q] = 0.f;
  for (int n = tid; n < NQ; n += blockDim.x) {
    int t = b * NQ + n;
    float w = (gt[t * 2 + 1] > gt[t * 2 + 0]) ? 1.f : 0.f;  // argmax(2) as float; w*w == w
    if (w == 0.f) continue;
    float z = lz[t];
    float X = xyz[t * 3 + 0] * z, Y = xyz[t * 3 + 1] * z, Z = xyz[t * 3 + 2] * z;
    float u = corres[t * 3 + 0], v = corres[t * 3 + 1];
    float r1[12] = {X, Y, Z, 1.f, 0.f, 0.f, 0.f, 0.f, -u * X, -u * Y, -u * Z, -u};
    float r2[12] = {0.f, 0.f, 0.f, 0.f, X, Y, Z, 1.f, -v * X, -v * Y, -v * Z, -v};
    int q = 0;
    for (int i = 0; i < 12; ++i)
      for (int j = i; j < 12; ++j, ++q)
        acc[q] += r1[i] * r1[j] + r2[i] * r2[j];
  }
  __shared__ float red[78];
  if (tid < 78) red[tid] = 0.f;
  __syncthreads();
  for (int q = 0; q < 78; ++q) atomicAdd(&red[q], acc[q]);
  __syncthreads();
  if (tid < 78) Macc[b * 78 + tid] = red[tid];
}

// ---------------------------------------------------------------------------
// PnP solve: 12x12 Jacobi eigh -> P ; sign fix ; 3x3 SVD via eigh(P^T P);
// R = U V^T ; t = P[:,3]/mean(S).  One lane per batch.
// ---------------------------------------------------------------------------
__device__ inline void jacobi3(const float Cin[3][3], float e[3], float Q[3][3]) {
  float A[3][3];
  for (int r = 0; r < 3; ++r)
    for (int c = 0; c < 3; ++c) { A[r][c] = Cin[r][c]; Q[r][c] = (r == c) ? 1.f : 0.f; }
  for (int sweep = 0; sweep < 15; ++sweep)
    for (int p = 0; p < 2; ++p)
      for (int q = p + 1; q < 3; ++q) {
        float apq = A[p][q];
        if (fabsf(apq) < 1e-20f) continue;
        float th = (A[q][q] - A[p][p]) / (2.f * apq);
        float tt = (th >= 0.f ? 1.f : -1.f) / (fabsf(th) + sqrtf(th * th + 1.f));
        float c = rsqrtf(tt * tt + 1.f), s = tt * c;
        for (int r = 0; r < 3; ++r) { float x = A[r][p], y = A[r][q]; A[r][p] = c * x - s * y; A[r][q] = s * x + c * y; }
        for (int r = 0; r < 3; ++r) { float x = A[p][r], y = A[q][r]; A[p][r] = c * x - s * y; A[q][r] = s * x + c * y; }
        for (int r = 0; r < 3; ++r) { float x = Q[r][p], y = Q[r][q]; Q[r][p] = c * x - s * y; Q[r][q] = s * x + c * y; }
      }
  for (int i = 0; i < 3; ++i) e[i] = A[i][i];
}

__global__ void pnp_solve_kernel(const float* __restrict__ Macc, float* __restrict__ out) {
  int b = blockIdx.x;
  if (threadIdx.x != 0) return;
  float A[12][12], V[12][12];
  {
    int q = 0;
    for (int i = 0; i < 12; ++i)
      for (int j = i; j < 12; ++j, ++q) { A[i][j] = Macc[b * 78 + q]; A[j][i] = A[i][j]; }
    for (int i = 0; i < 12; ++i)
      for (int j = 0; j < 12; ++j) V[i][j] = (i == j) ? 1.f : 0.f;
  }
  for (int sweep = 0; sweep < 30; ++sweep)
    for (int p = 0; p < 11; ++p)
      for (int q = p + 1; q < 12; ++q) {
        float apq = A[p][q];
        if (fabsf(apq) < 1e-18f) continue;
        float th = (A[q][q] - A[p][p]) / (2.f * apq);
        float tt = (th >= 0.f ? 1.f : -1.f) / (fabsf(th) + sqrtf(th * th + 1.f));
        float c = rsqrtf(tt * tt + 1.f), s = tt * c;
        for (int r = 0; r < 12; ++r) { float x = A[r][p], y = A[r][q]; A[r][p] = c * x - s * y; A[r][q] = s * x + c * y; }
        for (int r = 0; r < 12; ++r) { float x = A[p][r], y = A[q][r]; A[p][r] = c * x - s * y; A[q][r] = s * x + c * y; }
        for (int r = 0; r < 12; ++r) { float x = V[r][p], y = V[r][q]; V[r][p] = c * x - s * y; V[r][q] = s * x + c * y; }
      }
  int mi = 0;
  for (int i = 1; i < 12; ++i) if (A[i][i] < A[mi][mi]) mi = i;
  float P[3][4];
  for (int r = 0; r < 3; ++r)
    for (int c = 0; c < 4; ++c) P[r][c] = V[r * 4 + c][mi];
  float det = P[0][0] * (P[1][1] * P[2][2] - P[1][2] * P[2][1])
            - P[0][1] * (P[1][0] * P[2][2] - P[1][2] * P[2][0])
            + P[0][2] * (P[1][0] * P[2][1] - P[1][1] * P[2][0]);
  float sgn = (det >= 0.f) ? 1.f : -1.f;
  for (int r = 0; r < 3; ++r)
    for (int c = 0; c < 4; ++c) P[r][c] *= sgn;
  float C[3][3];
  for (int i = 0; i < 3; ++i)
    for (int j = 0; j < 3; ++j) {
      float a = 0.f;
      for (int r = 0; r < 3; ++r) a += P[r][i] * P[r][j];
      C[i][j] = a;
    }
  float e[3], V3[3][3];
  jacobi3(C, e, V3);
  float s0 = sqrtf(fmaxf(e[0], 0.f)), s1 = sqrtf(fmaxf(e[1], 0.f)), s2 = sqrtf(fmaxf(e[2], 0.f));
  float sv[3] = {s0, s1, s2};
  float U[3][3];
  for (int i = 0; i < 3; ++i) {
    float invs = 1.f / (sv[i] + 1e-20f);
    for (int r = 0; r < 3; ++r) {
      float a = 0.f;
      for (int k = 0; k < 3; ++k) a += P[r][k] * V3[k][i];
      U[r][i] = a * invs;
    }
  }
  for (int r = 0; r < 3; ++r)
    for (int c = 0; c < 3; ++c) {
      float a = 0.f;
      for (int k = 0; k < 3; ++k) a += U[r][k] * V3[c][k];
      out[b * 9 + r * 3 + c] = a;
    }
  float sm = (s0 + s1 + s2) * (1.f / 3.f);
  for (int r = 0; r < 3; ++r) out[18 + b * 3 + r] = P[r][3] / sm;
}

// ---------------------------------------------------------------------------
extern "C" void kernel_launch(void* const* d_in, const int* in_sizes, int n_in,
                              void* d_out, int out_size, void* d_ws, size_t ws_size,
                              hipStream_t stream) {
  (void)in_sizes; (void)n_in; (void)out_size; (void)ws_size;
  const float* xyz  = (const float*)d_in[0];   // (B,N,3)
  const float* pts  = (const float*)d_in[1];   // (B,N,64)
  const float* rf3  = (const float*)d_in[2];   // (B,64,H,W)
  const float* rf3i = (const float*)d_in[3];   // (B,3,H,W)
  const float* lz   = (const float*)d_in[4];   // (B,N,1)
  const float* gt   = (const float*)d_in[5];   // (B,N,2)
  float* out = (float*)d_out;                  // [R(18) | t(6) | weights(32768)]

  char* ws = (char*)d_ws;
  size_t off = 0;
  auto alloc = [&](size_t bytes) -> void* {
    void* p = (void*)(ws + off);
    off = (off + bytes + 255) & ~(size_t)255;
    return p;
  };

  const int Mr = BQ * NQ * KNN;  // 262144 rows (multiple of 128)
  int*      idxb = (int*)      alloc((size_t)Mr * sizeof(int));
  _Float16* X0   = (_Float16*) alloc((size_t)Mr * K0PAD * sizeof(_Float16));
  _Float16* Y1   = (_Float16*) alloc((size_t)Mr * N1 * sizeof(_Float16));
  _Float16* Y2   = (_Float16*) alloc((size_t)Mr * N2 * sizeof(_Float16));
  float*    Y3   = (float*)X0;   // alias: X0 dead after layer 1 (Mr*64*4 < Mr*160*2)
  _Float16* W1f  = (_Float16*) alloc((size_t)K0PAD * N1 * sizeof(_Float16));
  float*    b1f  = (float*)    alloc(N1 * sizeof(float));
  _Float16* W2f  = (_Float16*) alloc((size_t)N1 * N2 * sizeof(_Float16));
  float*    b2f  = (float*)    alloc(N2 * sizeof(float));
  _Float16* W3f  = (_Float16*) alloc((size_t)N2 * N3 * sizeof(_Float16));
  float*    b3f  = (float*)    alloc(N3 * sizeof(float));
  float*    G1   = (float*)    alloc(64 * 64 * sizeof(float));
  float*    g1b  = (float*)    alloc(64 * sizeof(float));
  float*    G2   = (float*)    alloc(128 * 64 * sizeof(float));
  float*    g2b  = (float*)    alloc(128 * sizeof(float));
  float*    G3   = (float*)    alloc(2 * 128 * sizeof(float));
  float*    g3b  = (float*)    alloc(2 * sizeof(float));
  float*    corres = (float*)  alloc((size_t)BQ * NQ * 3 * sizeof(float));
  float*    xfeat  = (float*)  alloc((size_t)BQ * NQ * 64 * sizeof(float));
  float*    Macc   = (float*)  alloc(BQ * 78 * sizeof(float));

  // ---- fold BN into weights (B-fragment-ordered f16 for WMMA layers) -------
  {
    int tot = K0PAD * N1;
    fold2d_kernel<<<(tot + 255) / 256, 256, 0, stream>>>(
        (const float*)d_in[6], (const float*)d_in[7], (const float*)d_in[8],
        (const float*)d_in[9], (const float*)d_in[10], (const float*)d_in[11],
        C0, K0PAD, N1, W1f, b1f);
    tot = N1 * N2;
    fold2d_kernel<<<(tot + 255) / 256, 256, 0, stream>>>(
        (const float*)d_in[12], (const float*)d_in[13], (const float*)d_in[14],
        (const float*)d_in[15], (const float*)d_in[16], (const float*)d_in[17],
        N1, N1, N2, W2f, b2f);
    tot = N2 * N3;
    fold2d_kernel<<<(tot + 255) / 256, 256, 0, stream>>>(
        (const float*)d_in[18], (const float*)d_in[19], (const float*)d_in[20],
        (const float*)d_in[21], (const float*)d_in[22], (const float*)d_in[23],
        N2, N2, N3, W3f, b3f);
    tot = 64 * 64;
    fold1d_kernel<<<(tot + 255) / 256, 256, 0, stream>>>(
        (const float*)d_in[24], (const float*)d_in[25], (const float*)d_in[26],
        (const float*)d_in[27], (const float*)d_in[28], (const float*)d_in[29],
        64, 64, 1, G1, g1b);
    tot = 64 * 128;
    fold1d_kernel<<<(tot + 255) / 256, 256, 0, stream>>>(
        (const float*)d_in[30], (const float*)d_in[31], (const float*)d_in[32],
        (const float*)d_in[33], (const float*)d_in[34], (const float*)d_in[35],
        64, 128, 1, G2, g2b);
    tot = 128 * 2;
    fold1d_kernel<<<(tot + 255) / 256, 256, 0, stream>>>(
        (const float*)d_in[36], (const float*)d_in[37], (const float*)d_in[38],
        (const float*)d_in[39], (const float*)d_in[40], (const float*)d_in[41],
        128, 2, 0, G3, g3b);
  }

  // ---- KNN + feature build --------------------------------------------------
  knn_kernel<<<(BQ * NQ) / 256, 256, 0, stream>>>(xyz, rf3i, idxb);
  feat_kernel<<<Mr / 256, 256, 0, stream>>>(xyz, pts, rf3, rf3i, lz, idxb, X0);

  // ---- WMMA MLP3 (weights staged via Tensor Data Mover) ---------------------
  gemm_bn_relu_kernel<K0PAD, N1, false><<<Mr / 128, 256, 0, stream>>>(X0, W1f, b1f, Y1, nullptr);
  gemm_bn_relu_kernel<N1, N2, false><<<Mr / 128, 256, 0, stream>>>(Y1, W2f, b2f, Y2, nullptr);
  gemm_bn_relu_kernel<N2, N3, true ><<<Mr / 128, 256, 0, stream>>>(Y2, W3f, b3f, nullptr, Y3);

  // ---- attention + conv1d MLP ----------------------------------------------
  attn_kernel<<<(BQ * NQ) / 256, 256, 0, stream>>>(Y3, rf3i, idxb, corres, xfeat);
  mlp4_kernel<<<(BQ * NQ) / 256, 256, 0, stream>>>(xfeat, G1, g1b, G2, g2b, G3, g3b, out + 24);

  // ---- PnP ------------------------------------------------------------------
  pnp_accum_kernel<<<BQ, 256, 0, stream>>>(xyz, lz, corres, gt, Macc);
  pnp_solve_kernel<<<BQ, 32, 0, stream>>>(Macc, out);
}
